// GAT_83099027243501
// MI455X (gfx1250) — matrix operand order repaired and checked by reading, not compiled, not run on hardware
//
#include <hip/hip_runtime.h>
#include <hip/hip_bf16.h>
#include <math.h>

typedef __attribute__((ext_vector_type(16))) _Float16 v16h;
typedef __attribute__((ext_vector_type(8)))  float    v8f;

#define Nn    50000
#define Ee    600000
#define Et    650000      // E + N self-loops
#define HID   128
#define HEADS 4
#define DPH   32          // HID / HEADS
#define CLS   10
#define KT    4           // K tiles of 32 (K = 128)
#define NEG_SLOPE 0.2f
#define ORD_NEG_INF 0x007FFFFFu   // f2ord(-inf)

// ---------- helpers ----------
__device__ __forceinline__ unsigned f2ord(float f) {
    unsigned u = __float_as_uint(f);
    return (u & 0x80000000u) ? ~u : (u | 0x80000000u);
}
__device__ __forceinline__ float ord2f(unsigned u) {
    return (u & 0x80000000u) ? __uint_as_float(u & 0x7FFFFFFFu)
                             : __uint_as_float(~u);
}
__device__ __forceinline__ void edge_sd(const long long* __restrict__ ei,
                                        int e, int& s, int& d) {
    if (e < Ee) { s = (int)ei[e]; d = (int)ei[Ee + e]; }
    else        { s = e - Ee; d = s; }           // self loop
}

// ---------- pack W[128 x wcols] (f32) into WMMA B-fragment layout (f16) ----------
// Bp[((nt*KT + kt)*32 + lane)*16 + i] = W[(kt*32 + (lane>=16?8:0) + (i&7) + (i>=8?16:0))*wcols + nt*16 + (lane&15)]
// OOB columns (layer 2: cols 10..15) packed as zero, so the GEMM needs no guard.
__global__ void pack_w_k(const float* __restrict__ W, _Float16* __restrict__ Bp,
                         int wcols, int ntiles)
{
    int t = blockIdx.x * blockDim.x + threadIdx.x;
    if (t >= ntiles * KT * 32 * 16) return;
    int i    =  t        & 15;
    int lane = (t >> 4)  & 31;
    int kt   = (t >> 9)  & (KT - 1);
    int nt   =  t >> 11;
    int col  = nt * 16 + (lane & 15);
    int k    = kt * 32 + (lane >= 16 ? 8 : 0) + (i & 7) + (i >= 8 ? 16 : 0);
    Bp[t] = (col < wcols) ? (_Float16)W[k * wcols + col] : (_Float16)0.0f;
}

// ---------- WMMA GEMM: out[M x ldo] = A[M x 128](f32) * packedB ----------
// One wave32 per 16x(16*NT) tile; K=128 as 4 steps of v_wmma_f32_16x16x32_f16,
// NT independent accumulator chains sharing one A fragment.
template <int NT>
__global__ __launch_bounds__(32)
void gemm_wmma_f16(const float* __restrict__ A, const v16h* __restrict__ Bp,
                   float* __restrict__ out, int lda, int ldo)
{
    const int lane = threadIdx.x;            // 0..31, full wave
    const int nloc = lane & 15;
    const bool hi  = lane >= 16;
    const int row  = blockIdx.x * 16 + nloc; // A row for this lane

    v8f acc[NT];
    #pragma unroll
    for (int t = 0; t < NT; ++t) acc[t] = (v8f){};

    #pragma unroll
    for (int kt = 0; kt < KT; ++kt) {
        // A fragment: 16-bit 16x32 layout (lanes 0-15: K +0..7 / +16..23; hi lanes +8)
        v16h a;
        const float* ap = A + row * lda + kt * 32 + (hi ? 8 : 0);
        #pragma unroll
        for (int i = 0; i < 8; ++i) {
            a[i]     = (_Float16)ap[i];
            a[8 + i] = (_Float16)ap[16 + i];
        }
        #pragma unroll
        for (int t = 0; t < NT; ++t) {
            int nt = blockIdx.y * NT + t;
            v16h b = Bp[(nt * KT + kt) * 32 + lane];   // contiguous 32B per lane
            acc[t] = __builtin_amdgcn_wmma_f32_16x16x32_f16(
                false, a, false, b, (short)0, acc[t], false, false);
        }
    }
    // C/D layout: VGPR r -> M = (hi?8:0)+r, N = lane%16
    #pragma unroll
    for (int t = 0; t < NT; ++t) {
        int colb = (blockIdx.y * NT + t) * 16 + nloc;
        #pragma unroll
        for (int r = 0; r < 8; ++r) {
            int m = (hi ? 8 : 0) + r;
            out[(blockIdx.x * 16 + m) * ldo + colb] = acc[t][r];
        }
    }
}

// ---------- attention scores: a_s[n,h], a_d[n,h] ----------
__global__ void att_scores_k(const float* __restrict__ hp,
                             const float* __restrict__ att_s,
                             const float* __restrict__ att_d,
                             float* __restrict__ as_o, float* __restrict__ ad_o,
                             int H, int D, int ldhp)
{
    int t = blockIdx.x * blockDim.x + threadIdx.x;
    if (t >= Nn * H) return;
    int n = t / H, h = t - n * H;
    const float* v  = hp + n * ldhp + h * D;
    const float* ws = att_s + h * D;
    const float* wd = att_d + h * D;
    float ss = 0.f, sd = 0.f;
    for (int d = 0; d < D; ++d) { float x = v[d]; ss += x * ws[d]; sd += x * wd[d]; }
    as_o[t] = ss; ad_o[t] = sd;
}

// ---------- fills ----------
__global__ void fill_u32_k(unsigned* p, unsigned v, int n) {
    int t = blockIdx.x * blockDim.x + threadIdx.x;
    if (t < n) p[t] = v;
}
__global__ void out_bias_k(float* __restrict__ out, const float* __restrict__ b) {
    int t = blockIdx.x * blockDim.x + threadIdx.x;
    if (t < Nn * CLS) out[t] = b[t % CLS];
}

// ---------- edge pass 1: segment max of leaky_relu(a_s[src]+a_d[dst]) ----------
__global__ void edge_max_k(const long long* __restrict__ ei,
                           const float* __restrict__ a_s, const float* __restrict__ a_d,
                           unsigned* __restrict__ m, int H)
{
    int t = blockIdx.x * blockDim.x + threadIdx.x;
    if (t >= Et * H) return;
    int e = t / H, h = t - e * H;
    int s, d; edge_sd(ei, e, s, d);
    float al = a_s[s * H + h] + a_d[d * H + h];
    al = al > 0.f ? al : NEG_SLOPE * al;
    atomicMax(&m[d * H + h], f2ord(al));
}

// ---------- edge pass 2: ex = exp(alpha - m[dst]); denom[dst] += ex ----------
__global__ void edge_exp_k(const long long* __restrict__ ei,
                           const float* __restrict__ a_s, const float* __restrict__ a_d,
                           const unsigned* __restrict__ m,
                           float* __restrict__ ex, float* __restrict__ denom, int H)
{
    int t = blockIdx.x * blockDim.x + threadIdx.x;
    if (t >= Et * H) return;
    int e = t / H, h = t - e * H;
    int s, d; edge_sd(ei, e, s, d);
    float al = a_s[s * H + h] + a_d[d * H + h];
    al = al > 0.f ? al : NEG_SLOPE * al;
    float mv = ord2f(m[d * H + h]);           // every node has a self-loop -> finite
    float x  = __expf(al - mv);
    ex[t] = x;
    atomicAdd(&denom[d * H + h], x);
}

// ---------- edge pass 3: agg[dst, f] += hp[src, f] * ex/denom ----------
__global__ void edge_aggr_k(const long long* __restrict__ ei,
                            const float* __restrict__ hp, const float* __restrict__ ex,
                            const float* __restrict__ denom, float* __restrict__ agg,
                            int H, int D, int ldhp, int ldo)
{
    int e = blockIdx.x;
    int f = threadIdx.x;
    if (f >= H * D) return;
    int s, d; edge_sd(ei, e, s, d);
    int h = f / D;
    float w = ex[e * H + h] / (denom[d * H + h] + 1e-16f);
    atomicAdd(&agg[d * ldo + f], hp[s * ldhp + f] * w);
}

// ---------- epilogue: h = elu(bn(agg + bias)) ----------
__global__ void epilogue_k(const float* __restrict__ agg, const float* __restrict__ bias,
                           const float* __restrict__ g, const float* __restrict__ be,
                           float* __restrict__ hout)
{
    int t = blockIdx.x * blockDim.x + threadIdx.x;
    if (t >= Nn * HID) return;
    int c = t & (HID - 1);
    float v = agg[t] + bias[c];
    v = v * (g[c] * rsqrtf(1.0f + 1e-5f)) + be[c];
    hout[t] = v > 0.f ? v : expm1f(v);
}

// ============================================================================
extern "C" void kernel_launch(void* const* d_in, const int* in_sizes, int n_in,
                              void* d_out, int out_size, void* d_ws, size_t ws_size,
                              hipStream_t stream)
{
    const float*     x  = (const float*)d_in[0];
    const long long* ei = (const long long*)d_in[1];
    const float *W0 = (const float*)d_in[2],  *as0 = (const float*)d_in[3],
                *ad0 = (const float*)d_in[4], *b0  = (const float*)d_in[5],
                *g0  = (const float*)d_in[6], *be0 = (const float*)d_in[7];
    const float *W1 = (const float*)d_in[8],  *as1 = (const float*)d_in[9],
                *ad1 = (const float*)d_in[10],*b1  = (const float*)d_in[11],
                *g1  = (const float*)d_in[12],*be1 = (const float*)d_in[13];
    const float *W2 = (const float*)d_in[14], *as2 = (const float*)d_in[15],
                *ad2 = (const float*)d_in[16],*b2  = (const float*)d_in[17];

    // ---- workspace carve-up (all L2-resident: ~91 MB total) ----
    char* base = (char*)d_ws;
    size_t off = 0;
    auto take = [&](size_t bytes) { char* p = base + off; off += (bytes + 255) & ~(size_t)255; return p; };
    float*    hp    = (float*)   take((size_t)Nn * HID * 4);   // GEMM out (layer2 uses Nn x 16)
    float*    agg   = (float*)   take((size_t)Nn * HID * 4);   // aggregation
    float*    hcur  = (float*)   take((size_t)Nn * HID * 4);   // post-activation h
    float*    a_s   = (float*)   take((size_t)Nn * HEADS * 4);
    float*    a_d   = (float*)   take((size_t)Nn * HEADS * 4);
    unsigned* mseg  = (unsigned*)take((size_t)Nn * HEADS * 4);
    float*    denom = (float*)   take((size_t)Nn * HEADS * 4);
    float*    exw   = (float*)   take((size_t)Et * HEADS * 4);
    _Float16* Bpack = (_Float16*)take((size_t)8 * KT * 32 * 16 * 2);  // packed W (max 8 n-tiles)
    float*    outp  = (float*)d_out;                            // [Nn x CLS]

    const int B = 256;
    const int PACK128 = 8 * KT * 32 * 16;   // threads to pack a 128x128 W
    const int PACK16  = 1 * KT * 32 * 16;   // threads to pack 128x10 -> one 16-wide tile
    dim3 gemm_grid (Nn / 16, HID / 32);     // 3125 x 4, NT=2
    dim3 gemm_grid2(Nn / 16, 1);            // layer 2: one padded 16-wide tile, NT=1

    // =================== Layer 0 ===================
    pack_w_k<<<(PACK128 + B-1)/B, B, 0, stream>>>(W0, Bpack, HID, 8);
    gemm_wmma_f16<2><<<gemm_grid, 32, 0, stream>>>(x, (const v16h*)Bpack, hp, HID, HID);
    att_scores_k<<<(Nn*HEADS + B-1)/B, B, 0, stream>>>(hp, as0, ad0, a_s, a_d, HEADS, DPH, HID);
    fill_u32_k<<<(Nn*HEADS + B-1)/B, B, 0, stream>>>(mseg, ORD_NEG_INF, Nn*HEADS);
    fill_u32_k<<<(Nn*HEADS + B-1)/B, B, 0, stream>>>((unsigned*)denom, 0u, Nn*HEADS);
    fill_u32_k<<<(Nn*HID   + B-1)/B, B, 0, stream>>>((unsigned*)agg,   0u, Nn*HID);
    edge_max_k <<<(Et*HEADS + B-1)/B, B, 0, stream>>>(ei, a_s, a_d, mseg, HEADS);
    edge_exp_k <<<(Et*HEADS + B-1)/B, B, 0, stream>>>(ei, a_s, a_d, mseg, exw, denom, HEADS);
    edge_aggr_k<<<Et, HID, 0, stream>>>(ei, hp, exw, denom, agg, HEADS, DPH, HID, HID);
    epilogue_k <<<(Nn*HID + B-1)/B, B, 0, stream>>>(agg, b0, g0, be0, hcur);

    // =================== Layer 1 ===================
    pack_w_k<<<(PACK128 + B-1)/B, B, 0, stream>>>(W1, Bpack, HID, 8);
    gemm_wmma_f16<2><<<gemm_grid, 32, 0, stream>>>(hcur, (const v16h*)Bpack, hp, HID, HID);
    att_scores_k<<<(Nn*HEADS + B-1)/B, B, 0, stream>>>(hp, as1, ad1, a_s, a_d, HEADS, DPH, HID);
    fill_u32_k<<<(Nn*HEADS + B-1)/B, B, 0, stream>>>(mseg, ORD_NEG_INF, Nn*HEADS);
    fill_u32_k<<<(Nn*HEADS + B-1)/B, B, 0, stream>>>((unsigned*)denom, 0u, Nn*HEADS);
    fill_u32_k<<<(Nn*HID   + B-1)/B, B, 0, stream>>>((unsigned*)agg,   0u, Nn*HID);
    edge_max_k <<<(Et*HEADS + B-1)/B, B, 0, stream>>>(ei, a_s, a_d, mseg, HEADS);
    edge_exp_k <<<(Et*HEADS + B-1)/B, B, 0, stream>>>(ei, a_s, a_d, mseg, exw, denom, HEADS);
    edge_aggr_k<<<Et, HID, 0, stream>>>(ei, hp, exw, denom, agg, HEADS, DPH, HID, HID);
    epilogue_k <<<(Nn*HID + B-1)/B, B, 0, stream>>>(agg, b1, g1, be1, hcur);

    // =================== Layer 2 (1 head, 10 classes, padded to 16) ===================
    pack_w_k<<<(PACK16 + B-1)/B, B, 0, stream>>>(W2, Bpack, CLS, 1);
    gemm_wmma_f16<1><<<gemm_grid2, 32, 0, stream>>>(hcur, (const v16h*)Bpack, hp, HID, 16);
    att_scores_k<<<(Nn + B-1)/B, B, 0, stream>>>(hp, as2, ad2, a_s, a_d, 1, CLS, 16);
    fill_u32_k<<<(Nn + B-1)/B, B, 0, stream>>>(mseg, ORD_NEG_INF, Nn);
    fill_u32_k<<<(Nn + B-1)/B, B, 0, stream>>>((unsigned*)denom, 0u, Nn);
    out_bias_k<<<(Nn*CLS + B-1)/B, B, 0, stream>>>(outp, b2);   // out = bias, then += messages
    edge_max_k <<<(Et + B-1)/B, B, 0, stream>>>(ei, a_s, a_d, mseg, 1);
    edge_exp_k <<<(Et + B-1)/B, B, 0, stream>>>(ei, a_s, a_d, mseg, exw, denom, 1);
    edge_aggr_k<<<Et, 32, 0, stream>>>(ei, hp, exw, denom, outp, 1, CLS, 16, CLS);
}
